// Attention_21749714386968
// MI455X (gfx1250) — matrix operand order repaired and checked by reading, not compiled, not run on hardware
//
#include <hip/hip_runtime.h>

// ---------------------------------------------------------------------------
// Fused multi-head attention + per-score MLP mask for MI455X (gfx1250, wave32)
// B=2, N=1024, L=4096, C=256, H=8, D=32.
// f16 WMMA (v_wmma_f32_16x16x32_f16) with f32 accumulation, flash-style
// single-pass softmax. K/V chunks are DMA'd into LDS with the Tensor Data
// Mover (tensor_load_to_lds + s_wait_tensorcnt), V pre-transposed per batch
// so every WMMA fragment load is a contiguous ds_load_b128.
// ---------------------------------------------------------------------------

typedef __attribute__((ext_vector_type(16))) _Float16 v16h;
typedef __attribute__((ext_vector_type(8)))  float    v8f;
typedef unsigned int v4u __attribute__((ext_vector_type(4)));
typedef unsigned int v8u __attribute__((ext_vector_type(8)));

__device__ __forceinline__ v8f wmma_f16(v16h a, v16h b, v8f c) {
  // 8 args: (neg_a, A, neg_b, B, c_mod, C, reuse_a, reuse_b)
  return __builtin_amdgcn_wmma_f32_16x16x32_f16(false, a, false, b,
                                                (short)0, c, false, false);
}

// 2-D TDM load: tile (tile_d0 x tile_d1) of 2-byte elements from a row-major
// tensor (tensor_d0 inner / tensor_d1 outer, stride0 elements between rows)
// into LDS at lds_addr. D# layout per CDNA5 ISA.
__device__ __forceinline__ void tdm_load_2d(unsigned lds_addr,
                                            unsigned long long gaddr,
                                            unsigned tensor_d0,
                                            unsigned tensor_d1,
                                            unsigned long long stride0,
                                            unsigned tile_d0,
                                            unsigned tile_d1) {
  v4u g0;
  g0[0] = 1u;                                        // count=1 (valid D#)
  g0[1] = lds_addr;                                  // lds_addr[31:0]
  g0[2] = (unsigned)(gaddr & 0xffffffffull);         // global_addr[31:0]
  g0[3] = (unsigned)((gaddr >> 32) & 0x01ffffffull)  // global_addr[56:32]
          | 0x80000000u;                             // type=2 ("image")
  v8u g1;
  g1[0] = 1u << 16;                                  // data_size=2B, mask=0
  g1[1] = (tensor_d0 & 0xffffu) << 16;               // tensor_dim0[15:0]
  g1[2] = ((tensor_d0 >> 16) & 0xffffu) | ((tensor_d1 & 0xffffu) << 16);
  g1[3] = ((tensor_d1 >> 16) & 0xffffu) | ((tile_d0 & 0xffffu) << 16);
  g1[4] = tile_d1 & 0xffffu;                         // tile_dim1, tile_dim2=0
  g1[5] = (unsigned)(stride0 & 0xffffffffull);       // dim0_stride[31:0]
  g1[6] = (unsigned)((stride0 >> 32) & 0xffffull);   // dim0_stride[47:32]
  g1[7] = 0u;                                        // dim1_stride unused (2D)
  asm volatile("tensor_load_to_lds %0, %1" :: "s"(g0), "s"(g1) : "memory");
}

// ---------------------------------------------------------------------------
// Projection GEMM: out[m,o] = sum_i in[m,i] * W[o,i] (+ bias[o])
// MODE 0: f16 row-major out. MODE 1: f32 out + bias.
// MODE 2: f16 out transposed per 4096-row batch: out[b][o][l] (for V).
// ---------------------------------------------------------------------------
enum { OUT_HALF = 0, OUT_FLOAT_BIAS = 1, OUT_HALF_T = 2 };

template <typename TIN, int MODE>
__global__ __launch_bounds__(256) void proj_gemm(const TIN* __restrict__ in,
                                                 const float* __restrict__ W,
                                                 const float* __restrict__ bias,
                                                 void* __restrict__ out) {
  const int lane = threadIdx.x & 31;
  const int wave = threadIdx.x >> 5;
  const int half = lane >> 4;
  const int l16  = lane & 15;
  const int row0 = blockIdx.x * 16;
  const int o0   = blockIdx.y * 128 + wave * 16;

  v8f acc = {};
  #pragma unroll 1
  for (int k0 = 0; k0 < 256; k0 += 32) {
    // A tile 16x32: lane row = l16, element e -> K = e + 16*half
    v16h a;
    const TIN* arow = in + (size_t)(row0 + l16) * 256 + k0 + 16 * half;
    #pragma unroll
    for (int j = 0; j < 8; ++j) {
      const int k = 2 * j;
      a[2 * j]     = (_Float16)arow[k];
      a[2 * j + 1] = (_Float16)arow[k + 1];
    }
    // B tile 32x16: B[k][n] = W[o0+n][k0+k]; lane col = l16, K = e + 16*half
    v16h bm;
    const float* wrow = W + (size_t)(o0 + l16) * 256 + k0 + 16 * half;
    #pragma unroll
    for (int e = 0; e < 16; ++e) bm[e] = (_Float16)wrow[e];

    acc = wmma_f16(a, bm, acc);
  }

  // D layout: element g -> row g + 8*half, col = l16
  if (MODE == OUT_HALF) {
    _Float16* o = (_Float16*)out;
    #pragma unroll
    for (int g = 0; g < 8; ++g)
      o[(size_t)(row0 + g + 8 * half) * 256 + o0 + l16] = (_Float16)acc[g];
  } else if (MODE == OUT_FLOAT_BIAS) {
    float* o = (float*)out;
    const float bv = bias ? bias[o0 + l16] : 0.f;
    #pragma unroll
    for (int g = 0; g < 8; ++g)
      o[(size_t)(row0 + g + 8 * half) * 256 + o0 + l16] = acc[g] + bv;
  } else {  // OUT_HALF_T: vT[b][c][l], batches of 4096 rows
    _Float16* o = (_Float16*)out;
    #pragma unroll
    for (int g = 0; g < 8; ++g) {
      const int m  = row0 + g + 8 * half;
      const int bb = m >> 12;        // batch
      const int l  = m & 4095;       // row within batch
      o[((size_t)(bb * 256 + o0 + l16)) * 4096 + l] = (_Float16)acc[g];
    }
  }
}

// ---------------------------------------------------------------------------
// Fused attention core. One block per (b, 16-row n tile); wave w = head w.
// Per 32-wide L chunk: TDM loads K[32x256] and Vt[256x32] into LDS ->
// 2 score WMMA -> LDS stage -> cross-head mask MLP -> online softmax
// (register stats, shfl_xor row reductions) -> 2 p@v WMMA.
// ---------------------------------------------------------------------------
__global__ __launch_bounds__(256) void attn_fused(
    const _Float16* __restrict__ qw, const _Float16* __restrict__ kw,
    const _Float16* __restrict__ vt,
    const float* __restrict__ W1, const float* __restrict__ b1,
    const float* __restrict__ W2, const float* __restrict__ b2,
    float* __restrict__ mask_out, _Float16* __restrict__ x_ws) {
  __shared__ _Float16 s_kt[32][256];   // K rows l0..l0+31, all channels (16KB)
  __shared__ _Float16 s_vt[256][32];   // Vt rows c, cols l0..l0+31     (16KB)
  __shared__ float    s_scores[8][16][32];  // raw scaled scores per head
  __shared__ _Float16 s_probs[8][16][32];   // softmax probs, row-major
  __shared__ float    s_mlp[81];            // W1(64) b1(8) W2(8) b2(1)

  const int tid = threadIdx.x;
  if (tid < 64)       s_mlp[tid] = W1[tid];
  else if (tid < 72)  s_mlp[tid] = b1[tid - 64];
  else if (tid < 80)  s_mlp[tid] = W2[tid - 72];
  else if (tid == 80) s_mlp[80]  = b2[0];

  const int lane = tid & 31;
  const int h    = tid >> 5;            // wave id == head id
  const int half = lane >> 4;
  const int l16  = lane & 15;
  const int b    = blockIdx.x >> 6;     // 64 n-tiles per batch
  const int n0   = (blockIdx.x & 63) * 16;

  // Q tile for this head in WMMA-A layout (16 rows x K=32)
  v16h qa;
  {
    const _Float16* qrow =
        qw + ((size_t)(b * 1024 + n0 + l16)) * 256 + h * 32 + 16 * half;
    #pragma unroll
    for (int j = 0; j < 8; ++j) {
      const int k = 2 * j;
      qa[2 * j]     = qrow[k];
      qa[2 * j + 1] = qrow[k + 1];
    }
  }

  float m_run[8], l_run[8];
  v8f accx0 = {}, accx1 = {};
  #pragma unroll
  for (int g = 0; g < 8; ++g) { m_run[g] = -1e30f; l_run[g] = 0.f; }

  const float scale = 0.17677669529663687f;  // 32^-0.5
  float* mrow = mask_out + ((size_t)(b * 1024 + n0)) * 4096;
  const unsigned lds_kt = (unsigned)(size_t)&s_kt[0][0];
  const unsigned lds_vt = (unsigned)(size_t)&s_vt[0][0];
  const unsigned long long kbase =
      (unsigned long long)(size_t)kw + (unsigned long long)b * 4096 * 256 * 2;
  const unsigned long long vbase =
      (unsigned long long)(size_t)vt + (unsigned long long)b * 256 * 4096 * 2;

  #pragma unroll 1
  for (int l0 = 0; l0 < 4096; l0 += 32) {
    __syncthreads();  // previous K/V tiles + s_scores fully consumed
    if (tid < 32) {   // wave 0 drives the Tensor Data Mover
      // K chunk: rows l0..l0+31 x 256 ch (row-major, stride 256 elems)
      tdm_load_2d(lds_kt, kbase + (unsigned long long)l0 * 512,
                  256u, 4096u, 256ull, 256u, 32u);
      // Vt chunk: 256 ch rows x cols l0..l0+31 (row-major, stride 4096 elems)
      tdm_load_2d(lds_vt, vbase + (unsigned long long)l0 * 2,
                  4096u, 256u, 4096ull, 32u, 256u);
      __builtin_amdgcn_s_wait_tensorcnt(0);
    }
    __syncthreads();  // tiles (and first-iter s_mlp) visible to all waves

    // ---- scores: S = Q (16x32) @ K^T, two 16x16 D tiles, from LDS ----
    v16h kb0, kb1;
    {
      const _Float16* kp0 = &s_kt[l16][h * 32 + 16 * half];
      const _Float16* kp1 = &s_kt[16 + l16][h * 32 + 16 * half];
      #pragma unroll
      for (int e = 0; e < 16; ++e) { kb0[e] = kp0[e]; kb1[e] = kp1[e]; }
    }
    v8f s0 = {}, s1 = {};
    s0 = wmma_f16(qa, kb0, s0);
    s1 = wmma_f16(qa, kb1, s1);
    #pragma unroll
    for (int g = 0; g < 8; ++g) { s0[g] *= scale; s1[g] *= scale; }

    // ---- stage raw scores for the cross-head mask MLP ----
    #pragma unroll
    for (int g = 0; g < 8; ++g) {
      s_scores[h][g + 8 * half][l16]      = s0[g];
      s_scores[h][g + 8 * half][16 + l16] = s1[g];
    }
    __syncthreads();

    // ---- mask MLP: relu(relu(s@W1^T+b1)@W2^T+b2), 512 entries / block ----
    #pragma unroll
    for (int r = 0; r < 2; ++r) {
      const int idx = tid * 2 + r;
      const int nl = idx >> 5, ll = idx & 31;
      float f[8];
      #pragma unroll
      for (int i = 0; i < 8; ++i) {
        float acc = s_mlp[64 + i];
        #pragma unroll
        for (int j = 0; j < 8; ++j) acc += s_mlp[i * 8 + j] * s_scores[j][nl][ll];
        f[i] = fmaxf(acc, 0.f);
      }
      float mo = s_mlp[80];
      #pragma unroll
      for (int i = 0; i < 8; ++i) mo += s_mlp[72 + i] * f[i];
      mrow[(size_t)nl * 4096 + l0 + ll] = fmaxf(mo, 0.f);
    }

    // ---- online softmax; D-layout rows reduce across 16-lane halves ----
    #pragma unroll
    for (int g = 0; g < 8; ++g) {
      float tmax = fmaxf(s0[g], s1[g]);
      #pragma unroll
      for (int m = 1; m < 16; m <<= 1) tmax = fmaxf(tmax, __shfl_xor(tmax, m, 32));
      const float mn   = fmaxf(m_run[g], tmax);
      const float corr = __expf(m_run[g] - mn);
      const float p0 = __expf(s0[g] - mn);
      const float p1 = __expf(s1[g] - mn);
      s0[g] = p0; s1[g] = p1;
      float ts = p0 + p1;
      #pragma unroll
      for (int m = 1; m < 16; m <<= 1) ts += __shfl_xor(ts, m, 32);
      l_run[g] = l_run[g] * corr + ts;
      m_run[g] = mn;
      accx0[g] *= corr;
      accx1[g] *= corr;
    }

    // ---- probs D-layout -> LDS row-major -> reload in WMMA-A layout ----
    #pragma unroll
    for (int g = 0; g < 8; ++g) {
      s_probs[h][g + 8 * half][l16]      = (_Float16)s0[g];
      s_probs[h][g + 8 * half][16 + l16] = (_Float16)s1[g];
    }
    asm volatile("s_wait_dscnt 0" ::: "memory");  // same-wave LDS RAW
    v16h pa;
    {
      const _Float16* prow = &s_probs[h][l16][16 * half];
      #pragma unroll
      for (int j = 0; j < 8; ++j) {
        const int k = 2 * j;
        pa[2 * j]     = prow[k];
        pa[2 * j + 1] = prow[k + 1];
      }
    }

    // ---- x += P (16x32) @ V: B tiles contiguous from transposed V in LDS ----
    v16h vb0, vb1;
    {
      const _Float16* vp0 = &s_vt[h * 32 + l16][16 * half];
      const _Float16* vp1 = &s_vt[h * 32 + 16 + l16][16 * half];
      #pragma unroll
      for (int e = 0; e < 16; ++e) { vb0[e] = vp0[e]; vb1[e] = vp1[e]; }
    }
    accx0 = wmma_f16(pa, vb0, accx0);
    accx1 = wmma_f16(pa, vb1, accx1);
  }

  // ---- normalize, write pre-projection x (f16) to workspace ----
  #pragma unroll
  for (int g = 0; g < 8; ++g) {
    const float inv = 1.0f / l_run[g];
    _Float16* xo =
        x_ws + ((size_t)(b * 1024 + n0 + g + 8 * half)) * 256 + h * 32;
    xo[l16]      = (_Float16)(accx0[g] * inv);
    xo[16 + l16] = (_Float16)(accx1[g] * inv);
  }
}

// ---------------------------------------------------------------------------
extern "C" void kernel_launch(void* const* d_in, const int* in_sizes, int n_in,
                              void* d_out, int out_size, void* d_ws,
                              size_t ws_size, hipStream_t stream) {
  (void)in_sizes; (void)n_in; (void)out_size; (void)ws_size;
  const float* query = (const float*)d_in[0];   // [2,1024,256]
  const float* key   = (const float*)d_in[1];   // [2,4096,256]
  const float* value = (const float*)d_in[2];   // [2,4096,256]
  // d_in[3] key_padding_mask, d_in[4] hw_lvl: unused by the reference math
  const float* Wq = (const float*)d_in[5];
  const float* Wk = (const float*)d_in[6];
  const float* Wv = (const float*)d_in[7];
  const float* Wp = (const float*)d_in[8];
  const float* bp = (const float*)d_in[9];
  const float* W1 = (const float*)d_in[10];
  const float* b1 = (const float*)d_in[11];
  const float* W2 = (const float*)d_in[12];
  const float* b2 = (const float*)d_in[13];

  // f16 workspace: q(1MB) k(4MB) vT(4MB) x_pre(1MB)
  _Float16* qws = (_Float16*)d_ws;
  _Float16* kws = qws + (size_t)2 * 1024 * 256;
  _Float16* vtw = kws + (size_t)2 * 4096 * 256;   // [b][c][l] transposed
  _Float16* xws = vtw + (size_t)2 * 4096 * 256;

  float* xout    = (float*)d_out;                    // [2,1024,256]
  float* maskout = xout + (size_t)2 * 1024 * 256;    // [2,1024,4096,1]

  const dim3 blk(256);
  proj_gemm<float, OUT_HALF><<<dim3(2048 / 16, 2), blk, 0, stream>>>(query, Wq, nullptr, qws);
  proj_gemm<float, OUT_HALF><<<dim3(8192 / 16, 2), blk, 0, stream>>>(key, Wk, nullptr, kws);
  proj_gemm<float, OUT_HALF_T><<<dim3(8192 / 16, 2), blk, 0, stream>>>(value, Wv, nullptr, vtw);

  attn_fused<<<dim3(128), blk, 0, stream>>>(qws, kws, vtw, W1, b1, W2, b2,
                                            maskout, xws);

  proj_gemm<_Float16, OUT_FLOAT_BIAS><<<dim3(2048 / 16, 2), blk, 0, stream>>>(xws, Wp, bp, xout);
}
